// SelfAttention_49709951484069
// MI455X (gfx1250) — matrix operand order repaired and checked by reading, not compiled, not run on hardware
//
#include <hip/hip_runtime.h>

// Problem constants (from the reference)
#define N_TOK   8192
#define D_IN    1024
#define D_OUT   128

typedef __attribute__((ext_vector_type(16))) __bf16 v16bf;
typedef __attribute__((ext_vector_type(8)))  __bf16 v8bf;
typedef __attribute__((ext_vector_type(8)))  float  v8f;

// softmax scale folded into Q at projection time, in log2 domain:
// (1/sqrt(128)) * log2(e)
#define QSCALE 0.12751744715940437f

// ---------------------------------------------------------------------------
// WMMA helpers (CDNA5 wave32, V_WMMA_F32_16X16X32_BF16)
// ---------------------------------------------------------------------------
__device__ __forceinline__ v8f wmma_bf16(v16bf a, v16bf b, v8f c) {
  return __builtin_amdgcn_wmma_f32_16x16x32_bf16(
      /*neg_a=*/false, a, /*neg_b=*/false, b,
      /*c_mod=*/(short)0, c, /*reuse_a=*/false, /*reuse_b=*/false);
}

// A-matrix 16x32 bf16 (M x K), row-major source with row stride `ld` (elems).
// ISA layout: lanes 0-15 -> M=lane, K={0..7,16..23}; lanes 16-31 -> M=lane-16,
// K={8..15,24..31}. Two 16B loads per lane. Works for global and LDS pointers.
template <typename P>
__device__ __forceinline__ v16bf load_a(P base, int ld, int lane) {
  int m  = lane & 15;
  int k0 = (lane < 16) ? 0 : 8;
  const __bf16* p = (const __bf16*)base + m * ld;
  v8bf lo = *(const v8bf*)(p + k0);
  v8bf hi = *(const v8bf*)(p + k0 + 16);
  v16bf a;
#pragma unroll
  for (int i = 0; i < 8; ++i) { a[i] = lo[i]; a[8 + i] = hi[i]; }
  return a;
}

// B-matrix 32x16 bf16 (K x N) where B[k][n] = src[n*ld + k] (src rows = N-dim,
// K contiguous). ISA layout: lanes 0-15 -> N=lane, K={0..7,16..23}; lanes 16-31 ->
// N=lane-16, K={8..15,24..31}. Two 16B loads per lane.
__device__ __forceinline__ v16bf load_b(const __bf16* __restrict__ base,
                                        int ld, int lane) {
  int n  = lane & 15;
  int k0 = (lane < 16) ? 0 : 8;
  const __bf16* p = base + n * ld;
  v8bf lo = *(const v8bf*)(p + k0);
  v8bf hi = *(const v8bf*)(p + k0 + 16);
  v16bf b;
#pragma unroll
  for (int i = 0; i < 8; ++i) { b[i] = lo[i]; b[8 + i] = hi[i]; }
  return b;
}

// ---------------------------------------------------------------------------
// Stage 1: f32 -> bf16 conversion
// ---------------------------------------------------------------------------
__global__ void cvt_bf16_kernel(const float* __restrict__ src,
                                __bf16* __restrict__ dst, int n) {
  int i = blockIdx.x * blockDim.x + threadIdx.x;
  int stride = gridDim.x * blockDim.x;
  for (; i < n; i += stride) dst[i] = (__bf16)src[i];
}

// ---------------------------------------------------------------------------
// Stage 2: projections Q = scale*(Xb*WqT), K = Xb*WkT, Vt = (Xb*WvT)^T
// One wave per 16x16 output tile; K-loop over D_IN in steps of 32.
// ---------------------------------------------------------------------------
__global__ __launch_bounds__(256)
void proj_kernel(const __bf16* __restrict__ Xb,
                 const __bf16* __restrict__ Wqb,
                 const __bf16* __restrict__ Wkb,
                 const __bf16* __restrict__ Wvb,
                 __bf16* __restrict__ Qb,
                 __bf16* __restrict__ Kb,
                 __bf16* __restrict__ Vtb) {
  int wave = (blockIdx.x * blockDim.x + threadIdx.x) >> 5;
  int lane = threadIdx.x & 31;
  // 3 matrices * (N_TOK/16) i-blocks * (D_OUT/16) o-blocks = 3*512*8 tiles
  int mat = wave >> 12;          // /4096
  int rem = wave & 4095;
  int i0  = (rem >> 3) * 16;     // token-row block
  int o0  = (rem & 7) * 16;      // out-feature block
  const __bf16* W = (mat == 0) ? Wqb : (mat == 1) ? Wkb : Wvb;

  v8f c = {0.f, 0.f, 0.f, 0.f, 0.f, 0.f, 0.f, 0.f};
#pragma unroll 4
  for (int d0 = 0; d0 < D_IN; d0 += 32) {
    v16bf a = load_a(Xb + i0 * D_IN + d0, D_IN, lane);   // X rows
    v16bf b = load_b(W  + o0 * D_IN + d0, D_IN, lane);   // W rows (= B cols)
    c = wmma_bf16(a, b, c);
  }

  // C layout: VGPR r -> row (r or r+8 by lane half), col = lane%16
  int col   = lane & 15;
  int mbase = (lane < 16) ? 0 : 8;
  if (mat == 2) {
#pragma unroll
    for (int r = 0; r < 8; ++r)
      Vtb[(o0 + col) * N_TOK + i0 + mbase + r] = (__bf16)c[r];
  } else if (mat == 0) {
#pragma unroll
    for (int r = 0; r < 8; ++r)   // fold softmax scale + log2(e) into Q
      Qb[(i0 + mbase + r) * D_OUT + o0 + col] = (__bf16)(c[r] * QSCALE);
  } else {
#pragma unroll
    for (int r = 0; r < 8; ++r)
      Kb[(i0 + mbase + r) * D_OUT + o0 + col] = (__bf16)c[r];
  }
}

// ---------------------------------------------------------------------------
// Stage 3: flash attention. One wave per 32-query block (two 16-row A tiles
// share every K/V B-operand -> halves the L2 K/V stream).
// ---------------------------------------------------------------------------
__device__ __forceinline__ void softmax_update(v8f s0, v8f s1,
                                               float* __restrict__ m,
                                               float* __restrict__ l,
                                               v8f* __restrict__ oc,
                                               __bf16 (*__restrict__ shP)[32],
                                               int lane) {
  int col   = lane & 15;
  int rbase = (lane < 16) ? 0 : 8;
#pragma unroll
  for (int r = 0; r < 8; ++r) {
    float x0 = s0[r];                 // already in log2 domain (scale in Q)
    float x1 = s1[r];
    float rm = fmaxf(x0, x1);
#pragma unroll
    for (int off = 1; off < 16; off <<= 1)
      rm = fmaxf(rm, __shfl_xor(rm, off, 32));
    float mn    = fmaxf(m[r], rm);
    float alpha = exp2f(m[r] - mn);   // 0 on first block (m = -inf)
    float p0 = exp2f(x0 - mn);
    float p1 = exp2f(x1 - mn);
    float rs = p0 + p1;
#pragma unroll
    for (int off = 1; off < 16; off <<= 1)
      rs += __shfl_xor(rs, off, 32);
    l[r] = l[r] * alpha + rs;
    m[r] = mn;
#pragma unroll
    for (int t = 0; t < 8; ++t) oc[t][r] *= alpha;   // rescale O row
    shP[rbase + r][col]      = (__bf16)p0;           // P -> LDS (C layout)
    shP[rbase + r][col + 16] = (__bf16)p1;
  }
}

__global__ __launch_bounds__(32)
void attn_kernel(const __bf16* __restrict__ Qb,
                 const __bf16* __restrict__ Kb,
                 const __bf16* __restrict__ Vtb,
                 float* __restrict__ out) {
  __shared__ __attribute__((aligned(32))) __bf16 shQ[32][D_OUT];  // 8 KB
  __shared__ __attribute__((aligned(32))) __bf16 shP[32][32];     // 2 KB
  int lane = threadIdx.x & 31;
  int i0   = blockIdx.x * 32;      // 256 blocks, 32 queries per wave

  // Stage Q tile (32 x 128) into LDS: one row per lane, 8 x 32B chunks
#pragma unroll
  for (int k = 0; k < 8; ++k)
    *(v16bf*)&shQ[lane][k * 16] =
        *(const v16bf*)(Qb + (i0 + lane) * D_OUT + k * 16);
  asm volatile("s_wait_dscnt 0" ::: "memory");

  v8f oc0[8], oc1[8];
#pragma unroll
  for (int t = 0; t < 8; ++t) {
    oc0[t] = (v8f){0.f, 0.f, 0.f, 0.f, 0.f, 0.f, 0.f, 0.f};
    oc1[t] = (v8f){0.f, 0.f, 0.f, 0.f, 0.f, 0.f, 0.f, 0.f};
  }
  float m0[8], l0[8], m1[8], l1[8];
#pragma unroll
  for (int r = 0; r < 8; ++r) {
    m0[r] = -__builtin_inff(); l0[r] = 0.f;
    m1[r] = -__builtin_inff(); l1[r] = 0.f;
  }

  for (int j0 = 0; j0 < N_TOK; j0 += 32) {
    // ---- S tiles: 32 queries x 32 keys; each B feeds two A's ----
    v8f s00 = {0.f,0.f,0.f,0.f,0.f,0.f,0.f,0.f};
    v8f s01 = s00, s10 = s00, s11 = s00;
#pragma unroll
    for (int t = 0; t < 4; ++t) {
      v16bf b0 = load_b(Kb + (j0     ) * D_OUT + t * 32, D_OUT, lane);
      v16bf b1 = load_b(Kb + (j0 + 16) * D_OUT + t * 32, D_OUT, lane);
      v16bf a0 = load_a(&shQ[0][t * 32],  D_OUT, lane);
      v16bf a1 = load_a(&shQ[16][t * 32], D_OUT, lane);
      s00 = wmma_bf16(a0, b0, s00);
      s01 = wmma_bf16(a0, b1, s01);
      s10 = wmma_bf16(a1, b0, s10);
      s11 = wmma_bf16(a1, b1, s11);
    }

    // ---- online softmax per 16-row block ----
    softmax_update(s00, s01, m0, l0, oc0, &shP[0],  lane);
    softmax_update(s10, s11, m1, l1, oc1, &shP[16], lane);
    asm volatile("s_wait_dscnt 0" ::: "memory");   // LDS store->load fence

    // ---- O += P (32x32) * V (32x128); each V B-operand feeds two A's ----
    v16bf ap0 = load_a(&shP[0][0],  32, lane);
    v16bf ap1 = load_a(&shP[16][0], 32, lane);
#pragma unroll
    for (int t = 0; t < 8; ++t) {
      v16bf bv = load_b(Vtb + (t * 16) * N_TOK + j0, N_TOK, lane);
      oc0[t] = wmma_bf16(ap0, bv, oc0[t]);
      oc1[t] = wmma_bf16(ap1, bv, oc1[t]);
    }
  }

  // ---- finalize: out = O / l ----
  int col   = lane & 15;
  int rbase = (lane < 16) ? 0 : 8;
#pragma unroll
  for (int t = 0; t < 8; ++t)
#pragma unroll
    for (int r = 0; r < 8; ++r) {
      out[(i0      + rbase + r) * D_OUT + t * 16 + col] = oc0[t][r] / l0[r];
      out[(i0 + 16 + rbase + r) * D_OUT + t * 16 + col] = oc1[t][r] / l1[r];
    }
}

// ---------------------------------------------------------------------------
// Launch
// ---------------------------------------------------------------------------
extern "C" void kernel_launch(void* const* d_in, const int* in_sizes, int n_in,
                              void* d_out, int out_size, void* d_ws, size_t ws_size,
                              hipStream_t stream) {
  const float* X  = (const float*)d_in[0];
  const float* Wq = (const float*)d_in[1];
  const float* Wk = (const float*)d_in[2];
  const float* Wv = (const float*)d_in[3];
  float* out = (float*)d_out;

  // Workspace layout (bytes)
  char* w = (char*)d_ws;
  __bf16* Xb  = (__bf16*)(w);                       // 8192*1024*2 = 16 MiB
  __bf16* Wqb = (__bf16*)(w + 16777216);            // 256 KiB
  __bf16* Wkb = (__bf16*)(w + 17039360);            // 256 KiB
  __bf16* Wvb = (__bf16*)(w + 17301504);            // 256 KiB
  __bf16* Qb  = (__bf16*)(w + 17563648);            // 2 MiB
  __bf16* Kb  = (__bf16*)(w + 19660800);            // 2 MiB
  __bf16* Vtb = (__bf16*)(w + 21757952);            // 2 MiB  (total ~22.8 MiB)

  // Stage 1: convert to bf16
  cvt_bf16_kernel<<<2048, 256, 0, stream>>>(X,  Xb,  N_TOK * D_IN);
  cvt_bf16_kernel<<<128,  256, 0, stream>>>(Wq, Wqb, D_OUT * D_IN);
  cvt_bf16_kernel<<<128,  256, 0, stream>>>(Wk, Wkb, D_OUT * D_IN);
  cvt_bf16_kernel<<<128,  256, 0, stream>>>(Wv, Wvb, D_OUT * D_IN);

  // Stage 2: projections. 3 * 512 * 8 = 12288 tiles, 8 waves per 256-thr block
  proj_kernel<<<1536, 256, 0, stream>>>(Xb, Wqb, Wkb, Wvb, Qb, Kb, Vtb);

  // Stage 3: flash attention, one wave per 32-query block
  attn_kernel<<<N_TOK / 32, 32, 0, stream>>>(Qb, Kb, Vtb, out);
}
